// Reconstructeur_68401649156500
// MI455X (gfx1250) — matrix operand order repaired and checked by reading, not compiled, not run on hardware
//
#include <hip/hip_runtime.h>
#include <hip/hip_bf16.h>

#define N_MLP   32
#define LATENT  1024
#define S0      512
#define S1      256
#define S2      128
#define OUT_DIM 3
#define NGS     64
#define GPTS    (NGS * NGS)       /* 4096 grid points */

#define TG      64                /* grid rows per block tile */
#define H0S     (S0 + 8)          /* LDS row stride (bf16 elems), padded */
#define H1S     (S1 + 8)
#define H2S     (S2 + 4)          /* f32 elems */

#define KSLAB   64                /* K-columns per staged weight slab */
#define SLW     (KSLAB + 8)       /* slab LDS row stride in bf16 (144 B) */
#define WBUF    (S1 * SLW * 2)    /* one slab buffer: 256 rows x 144 B = 36864 */

/* LDS byte offsets inside dynamic smem (dynamic segment starts at 0) */
#define OFF_H0  0
#define OFF_H1  (OFF_H0 + TG * H0S * 2)
#define OFF_H2  (OFF_H1 + TG * H1S * 2)
#define OFF_W   (OFF_H2 + TG * H2S * 4)
#define LDS_TOT (OFF_W + 2 * WBUF)

typedef __attribute__((ext_vector_type(16))) __bf16 v16bf;
typedef __attribute__((ext_vector_type(8)))  __bf16 v8bf;
typedef __attribute__((ext_vector_type(8)))  float  v8f;
typedef __attribute__((ext_vector_type(4)))  float  v4f;
typedef __attribute__((ext_vector_type(4)))  unsigned int u32x4;
typedef __attribute__((ext_vector_type(8)))  int   i32x8;
typedef __attribute__((ext_vector_type(4)))  int   i32x4;

#if defined(__has_builtin)
#if __has_builtin(__builtin_amdgcn_tensor_load_to_lds) && \
    __has_builtin(__builtin_amdgcn_s_wait_tensorcnt)
#define USE_TDM 1
#endif
#endif

// ---------------------------------------------------------------------------
// Stage one 2D bf16 weight slab (rows x KSLAB) into LDS.
//  TDM path: one tensor_load_to_lds descriptor (wave-uniform, SGPRs),
//            LDS rows padded to SLW via pad_interval=32dw, pad_amount=4dw.
//  Fallback: cooperative copy by the calling wave's 32 lanes.
// ---------------------------------------------------------------------------
__device__ __forceinline__ void stage_slab(const __bf16* gsrc, __bf16* ldst,
                                           unsigned lds_byte_off,
                                           unsigned rows, unsigned tensor_cols,
                                           unsigned row_stride_elems, int lane) {
#if defined(USE_TDM)
    (void)ldst; (void)lane;
    unsigned long long ga = (unsigned long long)(const void*)gsrc;
    u32x4 g0;
    g0[0] = 1u;                                   /* count=1, user descriptor */
    g0[1] = lds_byte_off;                         /* lds_addr                 */
    g0[2] = (unsigned)ga;                         /* global_addr[31:0]        */
    g0[3] = (unsigned)((ga >> 32) & 0x01FFFFFFu)  /* global_addr[56:32]       */
          | (2u << 30);                           /* type = 2 ("image")       */
    i32x8 g1;
    const unsigned td0 = tensor_cols;             /* tensor_dim0 (OOB bound)  */
    const unsigned td1 = rows;                    /* tensor_dim1              */
    const unsigned tile0 = KSLAB;                 /* tile_dim0 (contig elems) */
    const unsigned tile1 = rows;                  /* tile_dim1                */
    g1[0] = (int)((1u << 16)      /* data_size = 1 -> 2 bytes                */
                | (1u << 20)      /* pad_enable                              */
                | (4u << 22)      /* pad_interval code 4 -> every 32 dwords  */
                | (3u << 25));    /* pad_amount  code 3 -> 4 dwords (8 bf16) */
    g1[1] = (int)((td0 & 0xFFFFu) << 16);
    g1[2] = (int)(((td0 >> 16) & 0xFFFFu) | ((td1 & 0xFFFFu) << 16));
    g1[3] = (int)(((td1 >> 16) & 0xFFFFu) | (tile0 << 16));
    g1[4] = (int)(tile1 & 0xFFFFu);               /* tile_dim2 = 0 (2D)      */
    g1[5] = (int)row_stride_elems;                /* tensor_dim0_stride lo   */
    g1[6] = 0;                                    /* stride hi / dim1_stride */
    g1[7] = 0;
    i32x4 gz4 = (i32x4)0;                         /* groups 2/3 unused (2D)  */
    i32x8 gz8 = (i32x8)0;                         /* 6-arg toolchain extra   */
    __builtin_amdgcn_tensor_load_to_lds(g0, g1, gz4, gz4, gz8, 0);
#else
    (void)lds_byte_off;
    for (unsigned i = (unsigned)lane; i < rows * (KSLAB / 8); i += 32) {
        const unsigned r = i >> 3;
        const unsigned c = (i & 7u) * 8u;
        *(v8bf*)(ldst + (size_t)r * SLW + c) =
            *(const v8bf*)(gsrc + (size_t)r * row_stride_elems + c);
    }
#endif
}

__device__ __forceinline__ void wait_slab() {
#if defined(USE_TDM)
    __builtin_amdgcn_s_wait_tensorcnt(0);
#endif
}

// ---------------------------------------------------------------------------
// Prep 1: convert W1 (m,S1,S0) and W2 (m,S2,S1) f32 -> bf16 in workspace.
// ---------------------------------------------------------------------------
__global__ void k_convert_weights(const float* __restrict__ W1,
                                  const float* __restrict__ W2,
                                  __bf16* __restrict__ W1b,
                                  __bf16* __restrict__ W2b) {
    const int n1 = N_MLP * S1 * S0;
    const int n2 = N_MLP * S2 * S1;
    for (int i = blockIdx.x * blockDim.x + threadIdx.x; i < n1 + n2;
         i += gridDim.x * blockDim.x) {
        if (i < n1) W1b[i] = (__bf16)W1[i];
        else        W2b[i - n1] = (__bf16)W2[i - n1];
    }
}

// ---------------------------------------------------------------------------
// Prep 2: c0[m][o] = dot(Wa[m,o,:], x0) + ba[m,o] + bb[m,o]   (32 x 512)
// ---------------------------------------------------------------------------
__global__ __launch_bounds__(256) void k_c0(const float* __restrict__ x0,
                                            const float* __restrict__ Wa,
                                            const float* __restrict__ ba,
                                            const float* __restrict__ bb,
                                            float* __restrict__ c0) {
    __shared__ float sx[LATENT];
    const int m = blockIdx.x;
    for (int i = threadIdx.x; i < LATENT; i += blockDim.x) sx[i] = x0[i];
    __syncthreads();
    for (int o = threadIdx.x; o < S0; o += blockDim.x) {
        const float* w = Wa + ((size_t)m * S0 + o) * LATENT;
        float acc = 0.f;
        #pragma unroll 4
        for (int l = 0; l < LATENT; l += 4) {
            v4f wv = *(const v4f*)(w + l);
            acc += wv[0] * sx[l + 0] + wv[1] * sx[l + 1]
                 + wv[2] * sx[l + 2] + wv[3] * sx[l + 3];
        }
        c0[m * S0 + o] = acc + ba[m * S0 + o] + bb[m * S0 + o];
    }
}

// ---------------------------------------------------------------------------
// Main fused kernel: one block = (MLP m) x (64 grid rows).
//   TDM double-buffers W1/W2 K-slabs into LDS while WMMAs consume them.
// ---------------------------------------------------------------------------
__global__ __launch_bounds__(256) void k_mlp(
    const float*  __restrict__ grid,   // (G, 2)
    const float*  __restrict__ Wb,     // (m, S0, 2)
    const __bf16* __restrict__ W1b,    // (m, S1, S0) bf16
    const float*  __restrict__ b1,     // (m, S1)
    const __bf16* __restrict__ W2b,    // (m, S2, S1) bf16
    const float*  __restrict__ b2,     // (m, S2)
    const float*  __restrict__ W3,     // (m, 3, S2)
    const float*  __restrict__ b3,     // (m, 3)
    const float*  __restrict__ c0,     // (m, S0)  folded latent + biases
    float*        __restrict__ out)    // (m*G, 3)
{
    extern __shared__ char smem[];
    __bf16* sH0 = (__bf16*)(smem + OFF_H0);     // TG x H0S
    __bf16* sH1 = (__bf16*)(smem + OFF_H1);     // TG x H1S
    float*  sH2 = (float*) (smem + OFF_H2);     // TG x H2S

    const int m    = blockIdx.x;
    const int g0   = blockIdx.y * TG;
    const int tid  = threadIdx.x;
    const int lane = tid & 31;
    const int wave = tid >> 5;      // 0..7
    const int ln   = lane & 15;     // column / M selector
    const int hi   = lane >> 4;     // half-wave selector

    const __bf16* W1m = W1b + (size_t)m * S1 * S0;
    const __bf16* W2m = W2b + (size_t)m * S2 * S1;

    int cur = 0;
    // Kick off layer-1 slab 0 DMA before computing H0 (overlaps fully).
    if (wave == 0)
        stage_slab(W1m, (__bf16*)(smem + OFF_W), OFF_W, S1, S0, S0, lane);

    __builtin_prefetch(W3 + m * OUT_DIM * S2, 0, 1);   // warm tiny head weights

    // ---- Stage 0: H0[g][o] = relu(c0[o] + Wb0[o]*gx + Wb1[o]*gy) ----------
    {
        const float* c0m = c0 + m * S0;
        const float* Wbm = Wb + (size_t)m * S0 * 2;
        for (int i = tid; i < TG * S0; i += 256) {
            const int g = i >> 9;           // S0 == 512
            const int o = i & (S0 - 1);
            const float gx = grid[(g0 + g) * 2 + 0];
            const float gy = grid[(g0 + g) * 2 + 1];
            float v = c0m[o] + Wbm[o * 2 + 0] * gx + Wbm[o * 2 + 1] * gy;
            v = v > 0.f ? v : 0.f;
            sH0[g * H0S + o] = (__bf16)v;
        }
    }

    // ---- Layer 1: H1(64x256) = relu(H0(64x512) @ W1^T + b1) ---------------
    {
        const int r   = wave >> 1;          // row tile 0..3 (16 rows each)
        const int ct0 = (wave & 1) * 8;     // 8 column tiles per wave
        v8f acc[8] = {};
        const __bf16* arow = sH0 + (r * 16 + ln) * H0S;

        for (int ks = 0; ks < S0; ks += KSLAB) {
            if (wave == 0) wait_slab();     // slab[cur] landed in LDS
            __syncthreads();                // ...and everyone's past buf[1-cur]
            if (wave == 0 && ks + KSLAB < S0)
                stage_slab(W1m + ks + KSLAB,
                           (__bf16*)(smem + OFF_W + (1 - cur) * WBUF),
                           OFF_W + (unsigned)(1 - cur) * WBUF, S1, S0, S0, lane);

            const __bf16* bw = (const __bf16*)(smem + OFF_W + cur * WBUF);
            #pragma unroll
            for (int k2 = 0; k2 < KSLAB; k2 += 32) {
                const int kk = ks + k2;
                v8bf alo = *(const v8bf*)(arow + kk + hi * 16);
                v8bf ahi = *(const v8bf*)(arow + kk + hi * 16 + 8);
                v16bf a;
                #pragma unroll
                for (int j = 0; j < 8; ++j) { a[j] = alo[j]; a[j + 8] = ahi[j]; }

                #pragma unroll
                for (int t = 0; t < 8; ++t) {
                    const int pcol = (ct0 + t) * 16 + ln;
                    v16bf b = *(const v16bf*)(bw + pcol * SLW + k2 + hi * 16);
                    acc[t] = __builtin_amdgcn_wmma_f32_16x16x32_bf16(
                        false, a, false, b, (short)0, acc[t], false, false);
                }
            }
            cur ^= 1;
        }

        // Prefetch layer-2 slab 0 into the free buffer while epilogue runs.
        if (wave == 0)
            stage_slab(W2m, (__bf16*)(smem + OFF_W + cur * WBUF),
                       OFF_W + (unsigned)cur * WBUF, S2, S1, S1, lane);

        #pragma unroll
        for (int t = 0; t < 8; ++t) {
            const int pcol = (ct0 + t) * 16 + ln;
            const float bias = b1[m * S1 + pcol];
            #pragma unroll
            for (int v = 0; v < 8; ++v) {
                float x = acc[t][v] + bias;
                x = x > 0.f ? x : 0.f;
                sH1[(r * 16 + hi * 8 + v) * H1S + pcol] = (__bf16)x;
            }
        }
    }

    // ---- Layer 2: H2(64x128) = relu(H1(64x256) @ W2^T + b2) ---------------
    {
        const int r   = wave >> 1;
        const int ct0 = (wave & 1) * 4;     // 4 column tiles per wave
        v8f acc[4] = {};
        const __bf16* arow = sH1 + (r * 16 + ln) * H1S;

        for (int ks = 0; ks < S1; ks += KSLAB) {
            if (wave == 0) wait_slab();
            __syncthreads();                // also publishes sH1 on first pass
            if (wave == 0 && ks + KSLAB < S1)
                stage_slab(W2m + ks + KSLAB,
                           (__bf16*)(smem + OFF_W + (1 - cur) * WBUF),
                           OFF_W + (unsigned)(1 - cur) * WBUF, S2, S1, S1, lane);

            const __bf16* bw = (const __bf16*)(smem + OFF_W + cur * WBUF);
            #pragma unroll
            for (int k2 = 0; k2 < KSLAB; k2 += 32) {
                const int kk = ks + k2;
                v8bf alo = *(const v8bf*)(arow + kk + hi * 16);
                v8bf ahi = *(const v8bf*)(arow + kk + hi * 16 + 8);
                v16bf a;
                #pragma unroll
                for (int j = 0; j < 8; ++j) { a[j] = alo[j]; a[j + 8] = ahi[j]; }

                #pragma unroll
                for (int t = 0; t < 4; ++t) {
                    const int qcol = (ct0 + t) * 16 + ln;
                    v16bf b = *(const v16bf*)(bw + qcol * SLW + k2 + hi * 16);
                    acc[t] = __builtin_amdgcn_wmma_f32_16x16x32_bf16(
                        false, a, false, b, (short)0, acc[t], false, false);
                }
            }
            cur ^= 1;
        }

        #pragma unroll
        for (int t = 0; t < 4; ++t) {
            const int qcol = (ct0 + t) * 16 + ln;
            const float bias = b2[m * S2 + qcol];
            #pragma unroll
            for (int v = 0; v < 8; ++v) {
                float x = acc[t][v] + bias;
                x = x > 0.f ? x : 0.f;
                sH2[(r * 16 + hi * 8 + v) * H2S + qcol] = x;
            }
        }
    }
    __syncthreads();

    // ---- Layer 3: out(64x3) = tanh(H2 @ W3^T + b3), scalar VALU ----------
    {
        const float* W3m = W3 + m * OUT_DIM * S2;
        const float* b3m = b3 + m * OUT_DIM;
        for (int t = tid; t < TG * OUT_DIM; t += 256) {
            const int g = t / OUT_DIM;
            const int o = t % OUT_DIM;
            const float* h = sH2 + g * H2S;
            const float* w = W3m + o * S2;
            float acc = 0.f;
            #pragma unroll 8
            for (int q = 0; q < S2; ++q) acc += h[q] * w[q];
            out[((size_t)m * GPTS + g0 + g) * OUT_DIM + o] = tanhf(acc + b3m[o]);
        }
    }
}

// ---------------------------------------------------------------------------
extern "C" void kernel_launch(void* const* d_in, const int* in_sizes, int n_in,
                              void* d_out, int out_size, void* d_ws, size_t ws_size,
                              hipStream_t stream) {
    const float* x0   = (const float*)d_in[0];
    const float* grid = (const float*)d_in[1];
    const float* Wa   = (const float*)d_in[2];
    const float* ba   = (const float*)d_in[3];
    const float* Wb   = (const float*)d_in[4];
    const float* bb   = (const float*)d_in[5];
    const float* W1   = (const float*)d_in[6];
    const float* b1   = (const float*)d_in[7];
    const float* W2   = (const float*)d_in[8];
    const float* b2   = (const float*)d_in[9];
    const float* W3   = (const float*)d_in[10];
    const float* b3   = (const float*)d_in[11];
    float* out = (float*)d_out;

    char* ws = (char*)d_ws;
    const size_t w1b_bytes = (size_t)N_MLP * S1 * S0 * sizeof(__bf16);  // 8 MiB
    const size_t w2b_bytes = (size_t)N_MLP * S2 * S1 * sizeof(__bf16);  // 2 MiB
    __bf16* W1b = (__bf16*)ws;
    __bf16* W2b = (__bf16*)(ws + w1b_bytes);
    float*  c0  = (float*)(ws + w1b_bytes + w2b_bytes);

    k_convert_weights<<<1024, 256, 0, stream>>>(W1, W2, W1b, W2b);
    k_c0<<<N_MLP, 256, 0, stream>>>(x0, Wa, ba, bb, c0);

    dim3 gdim(N_MLP, GPTS / TG);   // 32 x 64 = 2048 blocks
    k_mlp<<<gdim, 256, (size_t)LDS_TOT, stream>>>(grid, Wb, W1b, b1, W2b, b2,
                                                  W3, b3, c0, out);
}
